// ProtTucker_55147380081181
// MI455X (gfx1250) — matrix operand order repaired and checked by reading, not compiled, not run on hardware
//
#include <hip/hip_runtime.h>
#include <hip/hip_bf16.h>
#include <math.h>

// ---------------------------------------------------------------------------
// Types for CDNA5 WMMA
// ---------------------------------------------------------------------------
typedef __bf16 bf16;
typedef __attribute__((ext_vector_type(16))) __bf16 v16bf;
typedef __attribute__((ext_vector_type(8)))  float  v8f;

// Problem constants
#define BATCH 256
#define DIM   1280
#define CCH   64            // C
#define WIN   14
#define NTOK  196           // WIN*WIN
#define NH    8
#define HD    8             // C / NH
#define NAG   49            // agent tokens
#define B3    768           // 3*BATCH

// ---------------------------------------------------------------------------
// CDNA5 async global->LDS staging (ISA 15.18.3, tracked by ASYNCcnt)
// ---------------------------------------------------------------------------
__device__ __forceinline__ void async_load_b128(unsigned lds_off, const void* gaddr) {
  asm volatile("global_load_async_to_lds_b128 %0, %1, off"
               :: "v"(lds_off), "v"(gaddr) : "memory");
}
__device__ __forceinline__ void wait_asynccnt0() {
  asm volatile("s_wait_asynccnt 0x0" ::: "memory");
}
__device__ __forceinline__ unsigned lds_offset_of(const void* p) {
  // __shared__ objects are allocated from LDS offset 0; low 32 bits of the
  // generic pointer are the wave-relative LDS byte offset.
  return (unsigned)(unsigned long long)p;
}

// ---------------------------------------------------------------------------
// Generic tiled GEMM:  C[M,N] = act( A[M,K] @ W[N,K]^T + bias[N] )
// A, W are bf16; accumulate f32 via v_wmma_f32_16x16x32_bf16.
// Block tile 128(M) x 64(N), K-step 32. 8 waves arranged 4(M) x 2(N);
// each wave owns 32x32 => 2 A-frags x 2 B-frags = 4 WMMAs per K-step.
// M%128==0, N%64==0, K%32==0 (all shapes in this model satisfy this).
// out_bf16: 0 -> f32 output, 1 -> bf16 output (for chained GEMMs).
// ---------------------------------------------------------------------------
#define BM 128
#define BN 64
#define BK 32
#define LSTR 40   // bf16 row stride: 80 B (16B-aligned rows, odd bank phase)

__global__ __launch_bounds__(256)
void gemm_bf16_wmma(const bf16* __restrict__ A, const bf16* __restrict__ W,
                    const float* __restrict__ bias, void* __restrict__ Cout,
                    int M, int N, int K, int act, int out_bf16) {
  __shared__ bf16 lds_a[BM * LSTR];   // 10 KB
  __shared__ bf16 lds_b[BN * LSTR];   // 5 KB

  const int t    = threadIdx.x;
  const int lane = t & 31;
  const int wave = t >> 5;
  const int wm   = wave >> 1;          // 0..3  (M dir, 32 rows each)
  const int wn   = wave & 1;           // 0..1  (N dir, 32 cols each)
  const int m_blk = blockIdx.y * BM;
  const int n_blk = blockIdx.x * BN;

  // A tile: 128x32 = 512 chunks of 8 bf16 -> 2 chunks/thread
  // B tile:  64x32 = 256 chunks of 8 bf16 -> 1 chunk/thread
  const int ar0 = t >> 1;                     // chunk t   : row t/2... (see below)
  // chunk c -> row c>>2, col8 (c&3)*8
  const int a0r = t >> 2,         a0c = (t & 3) * 8;
  const int a1r = (t + 256) >> 2, a1c = (t & 3) * 8;
  const int b0r = t >> 2,         b0c = (t & 3) * 8;
  (void)ar0;

  const unsigned la0 = lds_offset_of(&lds_a[a0r * LSTR + a0c]);
  const unsigned la1 = lds_offset_of(&lds_a[a1r * LSTR + a1c]);
  const unsigned lb0 = lds_offset_of(&lds_b[b0r * LSTR + b0c]);

  const bf16* ga0 = A + (size_t)(m_blk + a0r) * K + a0c;
  const bf16* ga1 = A + (size_t)(m_blk + a1r) * K + a1c;
  const bf16* gb0 = W + (size_t)(n_blk + b0r) * K + b0c;

  v8f acc00 = {}, acc01 = {}, acc10 = {}, acc11 = {};

  for (int k0 = 0; k0 < K; k0 += BK) {
    async_load_b128(la0, ga0 + k0);
    async_load_b128(la1, ga1 + k0);
    async_load_b128(lb0, gb0 + k0);
    wait_asynccnt0();
    __syncthreads();

    // ---- A fragments (16x32 bf16, ISA lane layout) ----
    v16bf af0, af1;
    {
      const int kb = (lane >> 4) * 8;
      const bf16* s0 = lds_a + (wm * 32 + (lane & 15)) * LSTR;
      const bf16* s1 = s0 + 16 * LSTR;
#pragma unroll
      for (int e = 0; e < 16; ++e) {
        const int p  = e >> 1;
        const int kk = kb + ((p & 4) ? 16 : 0) + (p & 3) * 2 + (e & 1);
        af0[e] = s0[kk];
        af1[e] = s1[kk];
      }
    }
    // ---- B fragments (32x16 bf16: lanes0-15 K=0..7,16..23, lanes16-31 K=8..15,24..31) ----
    v16bf bf0, bf1;
    {
      const int kb = (lane >> 4) * 8;
      const bf16* s0 = lds_b + (wn * 32 + (lane & 15)) * LSTR + kb;
      const bf16* s1 = s0 + 16 * LSTR;
#pragma unroll
      for (int e = 0; e < 16; ++e) { bf0[e] = s0[e + (e & 8)]; bf1[e] = s1[e + (e & 8)]; }
    }

    acc00 = __builtin_amdgcn_wmma_f32_16x16x32_bf16(false, af0, false, bf0,
                                                    (short)0, acc00, false, false);
    acc01 = __builtin_amdgcn_wmma_f32_16x16x32_bf16(false, af0, false, bf1,
                                                    (short)0, acc01, false, false);
    acc10 = __builtin_amdgcn_wmma_f32_16x16x32_bf16(false, af1, false, bf0,
                                                    (short)0, acc10, false, false);
    acc11 = __builtin_amdgcn_wmma_f32_16x16x32_bf16(false, af1, false, bf1,
                                                    (short)0, acc11, false, false);
    __syncthreads();
  }

  // ---- store: C/D layout — lane n=L&15, VGPR r -> m = r + (L>>4)*8 ----
  const int m0 = m_blk + wm * 32 + (lane >> 4) * 8;
  const int n0 = n_blk + wn * 32 + (lane & 15);
  const float bb0 = bias ? bias[n0]      : 0.0f;
  const float bb1 = bias ? bias[n0 + 16] : 0.0f;
  float* cf = (float*)Cout;
  bf16*  cb = (bf16*)Cout;
#pragma unroll
  for (int r = 0; r < 8; ++r) {
    float v00 = acc00[r] + bb0;                 // (m0+r,        n0)
    float v01 = acc01[r] + bb1;                 // (m0+r,        n0+16)
    float v10 = acc10[r] + bb0;                 // (m0+16+r,     n0)
    float v11 = acc11[r] + bb1;                 // (m0+16+r,     n0+16)
    if (act == 1) {
      v00 = tanhf(v00); v01 = tanhf(v01); v10 = tanhf(v10); v11 = tanhf(v11);
    }
    const size_t i0 = (size_t)(m0 + r) * N + n0;
    const size_t i1 = (size_t)(m0 + 16 + r) * N + n0;
    if (out_bf16) {
      cb[i0] = (bf16)v00;  cb[i0 + 16] = (bf16)v01;
      cb[i1] = (bf16)v10;  cb[i1 + 16] = (bf16)v11;
    } else {
      cf[i0] = v00;  cf[i0 + 16] = v01;
      cf[i1] = v10;  cf[i1 + 16] = v11;
    }
  }
}

// ---------------------------------------------------------------------------
// f32 -> bf16 conversion (one-time per launch; amortizes HBM bandwidth)
// ---------------------------------------------------------------------------
__global__ void cvt_f32_bf16_k(const float* __restrict__ src,
                               bf16* __restrict__ dst, int n) {
  int i = blockIdx.x * blockDim.x + threadIdx.x;
  if (i < n) dst[i] = (bf16)src[i];
}

// ---------------------------------------------------------------------------
// Positional bias precompute: bilinear 7x7 -> 14x14 (align_corners=False)
// ---------------------------------------------------------------------------
__device__ __forceinline__ float bil7(const float* s, int i, int j) {
  float si = 0.5f * i - 0.25f;  if (si < 0.f) si = 0.f;  if (si > 6.f) si = 6.f;
  float sj = 0.5f * j - 0.25f;  if (sj < 0.f) sj = 0.f;  if (sj > 6.f) sj = 6.f;
  int i0 = (int)floorf(si), j0 = (int)floorf(sj);
  int i1 = min(i0 + 1, 6),  j1 = min(j0 + 1, 6);
  float wi = si - (float)i0, wj = sj - (float)j0;
  float a = s[i0 * 7 + j0], b = s[i0 * 7 + j1];
  float c = s[i1 * 7 + j0], d = s[i1 * 7 + j1];
  return a * (1.f - wi) * (1.f - wj) + b * (1.f - wi) * wj
       + c * wi * (1.f - wj)        + d * wi * wj;
}

__global__ void bias_precompute(const float* __restrict__ an, const float* __restrict__ na,
                                const float* __restrict__ ah, const float* __restrict__ aw,
                                const float* __restrict__ ha, const float* __restrict__ wa,
                                float* __restrict__ pb, float* __restrict__ ab) {
  const int total = NH * NAG * NTOK;
  int idx = blockIdx.x * blockDim.x + threadIdx.x;
  if (idx >= 2 * total) return;
  const int which = idx / total;
  int r = idx - which * total;
  const int h = r / (NAG * NTOK); r -= h * NAG * NTOK;
  const int a = r / NTOK;
  const int n = r - a * NTOK;
  const int i = n / WIN, j = n - i * WIN;
  if (which == 0) {
    float v = bil7(an + (h * NAG + a) * 49, i, j)
            + ah[(h * NAG + a) * WIN + i] + aw[(h * NAG + a) * WIN + j];
    pb[(h * NAG + a) * NTOK + n] = v;
  } else {
    float v = bil7(na + (h * NAG + a) * 49, i, j)
            + ha[(h * WIN + i) * NAG + a] + wa[(h * WIN + j) * NAG + a];
    ab[(h * NTOK + n) * NAG + a] = v;
  }
}

// ---------------------------------------------------------------------------
// Agent pooling: 2x2 mean over q (first 64 channels of bf16 qkv) -> f32 agent
// ---------------------------------------------------------------------------
__global__ void agent_pool(const bf16* __restrict__ qkv, float* __restrict__ agent) {
  int idx = blockIdx.x * blockDim.x + threadIdx.x;
  const int total = B3 * NAG * CCH;
  if (idx >= total) return;
  const int b = idx / (NAG * CCH);
  int r = idx - b * NAG * CCH;
  const int a = r / CCH;
  const int c = r - a * CCH;
  const int ar = a / 7, ac = a - ar * 7;
  const bf16* base = qkv + (size_t)b * NTOK * (3 * CCH) + c;
  float s = 0.f;
#pragma unroll
  for (int dr = 0; dr < 2; ++dr)
#pragma unroll
    for (int dc = 0; dc < 2; ++dc) {
      const int n = (2 * ar + dr) * WIN + (2 * ac + dc);
      s += (float)base[(size_t)n * (3 * CCH)];
    }
  agent[((size_t)b * NAG + a) * CCH + c] = 0.25f * s;
}

// ---------------------------------------------------------------------------
// Agent attention, one (b,h) per block. Tiny K-dims (8/49) => VALU + LDS.
// ---------------------------------------------------------------------------
__global__ __launch_bounds__(256)
void agent_attention(const bf16* __restrict__ qkv, const float* __restrict__ agent,
                     const float* __restrict__ pb, const float* __restrict__ ab,
                     float* __restrict__ o) {
  __shared__ float sq[NTOK][HD];    // q * scale
  __shared__ float sk[NTOK][HD];
  __shared__ float sv[NTOK][HD];
  __shared__ float sag[NAG][HD];    // unscaled agent
  __shared__ float sav[NAG][HD];    // agent_v
  __shared__ float S[NAG * NTOK];   // stage1 [49][196] then stage2 [196][49]

  const int b = blockIdx.x;
  const int h = blockIdx.y;
  const int t = threadIdx.x;
  const float scale = 0.35355339059327373f;  // 8^-0.5

  const bf16* qbase = qkv + (size_t)b * NTOK * (3 * CCH) + h * HD;
  for (int idx = t; idx < NTOK * HD; idx += 256) {
    const int n = idx / HD, d = idx - n * HD;
    const size_t off = (size_t)n * (3 * CCH) + d;
    sq[n][d] = (float)qbase[off] * scale;
    sk[n][d] = (float)qbase[off + CCH];
    sv[n][d] = (float)qbase[off + 2 * CCH];
  }
  for (int idx = t; idx < NAG * HD; idx += 256) {
    const int a = idx / HD, d = idx - a * HD;
    sag[a][d] = agent[((size_t)b * NAG + a) * CCH + h * HD + d];
  }
  __syncthreads();

  // ---- stage 1: agents attend to tokens ----
  const float* pbh = pb + (size_t)h * NAG * NTOK;
  for (int idx = t; idx < NAG * NTOK; idx += 256) {
    const int a = idx / NTOK, n = idx - a * NTOK;
    float dot = 0.f;
#pragma unroll
    for (int d = 0; d < HD; ++d) dot += sag[a][d] * sk[n][d];
    S[idx] = dot * scale + pbh[idx];
  }
  __syncthreads();
  if (t < NAG) {                    // per-row softmax over 196
    float* row = S + t * NTOK;
    float mx = row[0];
    for (int n = 1; n < NTOK; ++n) mx = fmaxf(mx, row[n]);
    float sum = 0.f;
    for (int n = 0; n < NTOK; ++n) { float e = __expf(row[n] - mx); row[n] = e; sum += e; }
    const float inv = 1.f / sum;
    for (int n = 0; n < NTOK; ++n) row[n] *= inv;
  }
  __syncthreads();
  if (t < NAG * HD) {               // agent_v = attn @ v
    const int a = t / HD, d = t - a * HD;
    const float* row = S + a * NTOK;
    float acc = 0.f;
    for (int n = 0; n < NTOK; ++n) acc += row[n] * sv[n][d];
    sav[a][d] = acc;
  }
  __syncthreads();

  // ---- stage 2: tokens attend to agents ----
  const float* abh = ab + (size_t)h * NTOK * NAG;
  for (int idx = t; idx < NTOK * NAG; idx += 256) {
    const int n = idx / NAG, a = idx - n * NAG;
    float dot = 0.f;
#pragma unroll
    for (int d = 0; d < HD; ++d) dot += sq[n][d] * sag[a][d];
    S[idx] = dot + abh[idx];
  }
  __syncthreads();
  if (t < NTOK) {                   // per-row softmax over 49
    float* row = S + t * NAG;
    float mx = row[0];
    for (int a = 1; a < NAG; ++a) mx = fmaxf(mx, row[a]);
    float sum = 0.f;
    for (int a = 0; a < NAG; ++a) { float e = __expf(row[a] - mx); row[a] = e; sum += e; }
    const float inv = 1.f / sum;
    for (int a = 0; a < NAG; ++a) row[a] *= inv;
  }
  __syncthreads();
  for (int idx = t; idx < NTOK * HD; idx += 256) {
    const int n = idx / HD, d = idx - n * HD;
    const float* row = S + n * NAG;
    float acc = 0.f;
    for (int a = 0; a < NAG; ++a) acc += row[a] * sav[a][d];
    o[((size_t)b * NTOK + n) * CCH + h * HD + d] = acc;
  }
}

// ---------------------------------------------------------------------------
// Depthwise 3x3 conv on v, added into o; emits bf16 copy for the proj GEMM
// ---------------------------------------------------------------------------
__global__ void dwc_add(const bf16* __restrict__ qkv, const float* __restrict__ w,
                        const float* __restrict__ bias, const float* __restrict__ o,
                        bf16* __restrict__ ob) {
  int idx = blockIdx.x * blockDim.x + threadIdx.x;
  const int total = B3 * NTOK * CCH;
  if (idx >= total) return;
  const int b = idx / (NTOK * CCH);
  int r = idx - b * NTOK * CCH;
  const int n = r / CCH;
  const int c = r - n * CCH;
  const int y = n / WIN, x = n - y * WIN;
  const bf16* vbase = qkv + (size_t)b * NTOK * (3 * CCH) + 2 * CCH + c;
  const float* wc = w + c * 9;
  float acc = bias[c];
#pragma unroll
  for (int dy = 0; dy < 3; ++dy) {
    const int yy = y + dy - 1;
    if (yy < 0 || yy >= WIN) continue;
#pragma unroll
    for (int dx = 0; dx < 3; ++dx) {
      const int xx = x + dx - 1;
      if (xx < 0 || xx >= WIN) continue;
      acc += (float)vbase[(size_t)(yy * WIN + xx) * (3 * CCH)] * wc[dy * 3 + dx];
    }
  }
  ob[idx] = (bf16)(o[idx] + acc);
}

// ---------------------------------------------------------------------------
// Scatter MLP outputs: mlp2[(b*3+t)*128+j] -> out[t*256*128 + b*128 + j]
// ---------------------------------------------------------------------------
__global__ void scatter_triplet(const float* __restrict__ mlp2, float* __restrict__ out) {
  int idx = blockIdx.x * blockDim.x + threadIdx.x;
  const int total = B3 * 128;
  if (idx >= total) return;
  const int r = idx / 128, j = idx - r * 128;
  const int b = r / 3, tt = r - b * 3;
  out[tt * (BATCH * 128) + b * 128 + j] = mlp2[idx];
}

// ---------------------------------------------------------------------------
// Host-side orchestration
// ---------------------------------------------------------------------------
static inline dim3 gemm_grid(int M, int N) { return dim3(N / BN, M / BM); }
static inline void cvt(const float* s, bf16* d, int n, hipStream_t st) {
  cvt_f32_bf16_k<<<(n + 255) / 256, 256, 0, st>>>(s, d, n);
}

extern "C" void kernel_launch(void* const* d_in, const int* in_sizes, int n_in,
                              void* d_out, int out_size, void* d_ws, size_t ws_size,
                              hipStream_t stream) {
  const float* X       = (const float*)d_in[0];   // (256,3,1280)
  const float* W_embed = (const float*)d_in[1];   // (12544,1280)
  const float* b_embed = (const float*)d_in[2];
  const float* W_qkv   = (const float*)d_in[3];   // (192,64)
  const float* W_proj  = (const float*)d_in[4];   // (64,64)
  const float* b_proj  = (const float*)d_in[5];
  const float* dwc_w   = (const float*)d_in[6];   // (64,1,3,3)
  const float* dwc_b   = (const float*)d_in[7];
  const float* an_bias = (const float*)d_in[8];   // (8,49,7,7)
  const float* na_bias = (const float*)d_in[9];
  const float* ah_bias = (const float*)d_in[10];
  const float* aw_bias = (const float*)d_in[11];
  const float* ha_bias = (const float*)d_in[12];
  const float* wa_bias = (const float*)d_in[13];
  const float* W_adj   = (const float*)d_in[14];  // (256,12544)
  const float* b_adj   = (const float*)d_in[15];
  const float* W_t1    = (const float*)d_in[16];  // (256,1280)
  const float* b_t1    = (const float*)d_in[17];
  const float* W_t2    = (const float*)d_in[18];  // (128,256)
  const float* b_t2    = (const float*)d_in[19];
  float* out = (float*)d_out;
  (void)in_sizes; (void)n_in; (void)out_size; (void)ws_size;

  // ---- workspace layout (16B-aligned regions) ----
  char* w8 = (char*)d_ws;
  size_t off = 0;
  auto take = [&](size_t bytes) { char* p = w8 + off; off += (bytes + 15) & ~(size_t)15; return p; };

  bf16* Xb      = (bf16*)take((size_t)B3 * DIM * 2);                  // X bf16
  bf16* Wemb_b  = (bf16*)take((size_t)12544 * DIM * 2);
  bf16* Wqkv_b  = (bf16*)take((size_t)192 * 64 * 2);
  bf16* Wproj_b = (bf16*)take((size_t)64 * 64 * 2);
  bf16* Wadj_b  = (bf16*)take((size_t)256 * 12544 * 2);
  bf16* Wt1_b   = (bf16*)take((size_t)256 * DIM * 2);
  bf16* Wt2_b   = (bf16*)take((size_t)128 * 256 * 2);
  bf16* Xe_b    = (bf16*)take((size_t)B3 * 12544 * 2);                // embed out
  bf16* qkv_b   = (bf16*)take((size_t)B3 * NTOK * 3 * CCH * 2);       // qkv
  float* o_f    = (float*)take((size_t)B3 * NTOK * CCH * 4);          // attn out
  bf16* ob      = (bf16*)take((size_t)B3 * NTOK * CCH * 2);           // o + dwc
  bf16* oproj_b = (bf16*)take((size_t)B3 * NTOK * CCH * 2);           // proj out
  float* agent  = (float*)take((size_t)B3 * NAG * CCH * 4);
  float* pb     = (float*)take((size_t)NH * NAG * NTOK * 4);
  float* ab     = (float*)take((size_t)NH * NTOK * NAG * 4);
  bf16* mlp1_b  = (bf16*)take((size_t)B3 * 256 * 2);
  float* mlp2_f = (float*)take((size_t)B3 * 128 * 4);

  // 0) one-time f32 -> bf16 conversions (weights + X)
  cvt(X,       Xb,      B3 * DIM,      stream);
  cvt(W_embed, Wemb_b,  12544 * DIM,   stream);
  cvt(W_qkv,   Wqkv_b,  192 * 64,      stream);
  cvt(W_proj,  Wproj_b, 64 * 64,       stream);
  cvt(W_adj,   Wadj_b,  256 * 12544,   stream);
  cvt(W_t1,    Wt1_b,   256 * DIM,     stream);
  cvt(W_t2,    Wt2_b,   128 * 256,     stream);

  // 1) positional biases (bilinear 7->14 + decomposed biases)
  {
    const int total = 2 * NH * NAG * NTOK;
    bias_precompute<<<(total + 255) / 256, 256, 0, stream>>>(
        an_bias, na_bias, ah_bias, aw_bias, ha_bias, wa_bias, pb, ab);
  }

  // 2) embed: Xe(768,12544) = X @ W_embed^T + b_embed   (bf16 out)
  gemm_bf16_wmma<<<gemm_grid(B3, 12544), 256, 0, stream>>>(
      Xb, Wemb_b, b_embed, Xe_b, B3, 12544, DIM, 0, 1);

  // 3) qkv: (150528,192) = Xe_view(150528,64) @ W_qkv^T (no bias, bf16 out)
  gemm_bf16_wmma<<<gemm_grid(B3 * NTOK, 3 * CCH), 256, 0, stream>>>(
      Xe_b, Wqkv_b, nullptr, qkv_b, B3 * NTOK, 3 * CCH, CCH, 0, 1);

  // 4) agent tokens (2x2 mean pool of q)
  {
    const int total = B3 * NAG * CCH;
    agent_pool<<<(total + 255) / 256, 256, 0, stream>>>(qkv_b, agent);
  }

  // 5) two-stage agent attention -> o (f32)
  agent_attention<<<dim3(B3, NH), 256, 0, stream>>>(qkv_b, agent, pb, ab, o_f);

  // 6) depthwise conv enhancement: ob = bf16(o + dwc(v))
  {
    const int total = B3 * NTOK * CCH;
    dwc_add<<<(total + 255) / 256, 256, 0, stream>>>(qkv_b, dwc_w, dwc_b, o_f, ob);
  }

  // 7) proj: (150528,64) = ob @ W_proj^T + b_proj  (bf16 out)
  gemm_bf16_wmma<<<gemm_grid(B3 * NTOK, CCH), 256, 0, stream>>>(
      ob, Wproj_b, b_proj, oproj_b, B3 * NTOK, CCH, CCH, 0, 1);

  // 8) X_adj: (768,256) = oproj_view(768,12544) @ W_adj^T + b_adj -> d_out tail
  gemm_bf16_wmma<<<gemm_grid(B3, 256), 256, 0, stream>>>(
      oproj_b, Wadj_b, b_adj, out + 3 * BATCH * 128, B3, 256, 12544, 0, 0);

  // 9) triplet MLP: tanh(X @ W_t1^T + b_t1) @ W_t2^T + b_t2
  gemm_bf16_wmma<<<gemm_grid(B3, 256), 256, 0, stream>>>(
      Xb, Wt1_b, b_t1, mlp1_b, B3, 256, DIM, 1, 1);
  gemm_bf16_wmma<<<gemm_grid(B3, 128), 256, 0, stream>>>(
      mlp1_b, Wt2_b, b_t2, mlp2_f, B3, 128, 256, 0, 0);

  // 10) scatter anchor/pos/neg
  {
    const int total = B3 * 128;
    scatter_triplet<<<(total + 255) / 256, 256, 0, stream>>>(mlp2_f, out);
  }
}
